// AWLoss_86577950753131
// MI455X (gfx1250) — matrix operand order repaired and checked by reading, not compiled, not run on hardware
//
#include <hip/hip_runtime.h>
#include <hip/hip_bf16.h>

typedef __attribute__((ext_vector_type(16))) _Float16 v16h;
typedef __attribute__((ext_vector_type(8)))  float    v8f;

#define NFFT 511
#define NPAD 512
#define NSRC 256
#define PADL 127
#define NCH  48
#define AW_EPS  1e-9f
#define PI2  6.28318530717958647692f

// ---------------- workspace layout (bytes) ----------------
static const size_t SZ_A  = (size_t)NPAD * NSRC * sizeof(_Float16); // 256 KB (DFT matrix, row-major 512x256)
static const size_t SZ_AT = (size_t)NSRC * NPAD * sizeof(_Float16); // 256 KB (A^T, B-swizzled, K=256 N=512)
static const size_t SZ_X  = (size_t)NSRC * NSRC * sizeof(_Float16); // 128 KB (B-swizzled, K=256 N=256)
static const size_t SZ_G  = (size_t)NPAD * NSRC * sizeof(_Float16); // 256 KB (row-major 512x256)

// ---------------- WMMA fragment helpers (contiguous K order k = k0 + 16*half + j, same in A and B) ----------------
// A-matrix 16x32 f16 (row-major source): lane = (m&15) + 16*half;
// elems 0..15 -> K = k0 + half*16 + j
// => one contiguous 32B run per lane (2x global_load_b128).
__device__ __forceinline__ v16h load_a_frag(const _Float16* __restrict__ A, int ld,
                                            int row0, int k0, int lane) {
    const int m = row0 + (lane & 15);
    const int h = lane >> 4;
    const _Float16* p = A + (size_t)m * ld + k0 + h * 16;
    v16h a;
#pragma unroll
    for (int j = 0; j < 8; ++j) { a[j] = p[j]; a[j + 8] = p[8 + j]; }
    return a;
}

// B-matrix 32x16 f16 from K-block-swizzled storage:
//   Bs[(kb*N + n)*32 + t] = B[kb*32 + t][n]
// Fragment need: b[j] = B[k0 + 16h + j][col0 + (lane&15)]  (j = 0..15)
// => one contiguous 32B run per lane (2x global_load_b128, no packing movs).
__device__ __forceinline__ v16h load_b_swz(const _Float16* __restrict__ Bs, int N,
                                           int kb, int col0, int lane) {
    const _Float16* p = Bs + (((size_t)kb * N + col0 + (lane & 15)) << 5) + ((lane >> 4) << 4);
    v16h b;
#pragma unroll
    for (int j = 0; j < 16; ++j) b[j] = p[j];
    return b;
}

#define WMMA_F16(a, b, c) \
    __builtin_amdgcn_wmma_f32_16x16x32_f16(false, (a), false, (b), (short)0, (c), false, false)

// ---------------- kernels ----------------
__global__ void k_zero(float* p, int n) {
    int i = blockIdx.x * blockDim.x + threadIdx.x;
    if (i < n) p[i] = 0.0f;
}

// Build DFT matrix A[k][r] = exp(-2*pi*i * k*(r+127)/511) as f16 (row 511 zeroed):
//  - row-major Ar/Ai (stage-1 A operand, 512x256)
//  - B-swizzled transpose Atrs/Atis (stage-2 B operand: K=256 (r), N=512 (k))
// Exact integer phase reduction (no large-argument sincos error).
__global__ void k_twiddle(_Float16* __restrict__ Ar, _Float16* __restrict__ Ai,
                          _Float16* __restrict__ Atrs, _Float16* __restrict__ Atis) {
    int idx = blockIdx.x * blockDim.x + threadIdx.x;
    if (idx >= NPAD * NSRC) return;
    int k = idx >> 8;        // 0..511  (frequency, N dim of A^T)
    int r = idx & 255;       // 0..255  (source index, K dim of A^T)
    float cr = 0.0f, ci = 0.0f;
    if (k < NFFT) {
        int p = (k * (r + PADL)) % NFFT;           // exact, < 2^18
        float th = (PI2 / (float)NFFT) * (float)p;
        float s, c;
        sincosf(th, &s, &c);
        cr = c;
        ci = -s;
    }
    Ar[idx] = (_Float16)cr;
    Ai[idx] = (_Float16)ci;
    size_t swz = (((size_t)(r >> 5) * NPAD + k) << 5) + (r & 31);
    Atrs[swz] = (_Float16)cr;
    Atis[swz] = (_Float16)ci;
}

// Convert f32 channel images to f16 in B-swizzled layout (K=row r, N=col n).
__global__ void k_convert(const float* __restrict__ x, const float* __restrict__ y,
                          _Float16* __restrict__ Xs, _Float16* __restrict__ Ys) {
    int i = blockIdx.x * blockDim.x + threadIdx.x;
    if (i < NSRC * NSRC) {
        int r = i >> 8;
        int n = i & 255;
        size_t swz = (((size_t)(r >> 5) * NSRC + n) << 5) + (r & 31);
        Xs[swz] = (_Float16)x[i];
        Ys[swz] = (_Float16)y[i];
    }
}

// Stage 1: G = A_dft * D for D in {X(target), Y(recon)}; complex*real =>
// Gxr = Ar*X, Gxi = Ai*X, Gyr = Ar*Y, Gyi = Ai*Y.  Output 512x256 f16 row-major.
// 512 waves: 32 m-tiles x 16 n-tiles, K=256 in 8 steps, 4 WMMA/step.
__global__ void __launch_bounds__(256) k_stage1(
        const _Float16* __restrict__ Ar, const _Float16* __restrict__ Ai,
        const _Float16* __restrict__ Xs, const _Float16* __restrict__ Ys,
        _Float16* __restrict__ Gxr, _Float16* __restrict__ Gxi,
        _Float16* __restrict__ Gyr, _Float16* __restrict__ Gyi) {
    const int lane = threadIdx.x & 31;
    const int w = blockIdx.x * (blockDim.x >> 5) + (threadIdx.x >> 5);
    const int row0 = (w >> 4) * 16;   // m-tile (k1)
    const int col0 = (w & 15) * 16;   // n-tile (source column)
    v8f cxr = {}, cxi = {}, cyr = {}, cyi = {};
#pragma unroll
    for (int k0 = 0; k0 < NSRC; k0 += 32) {
        v16h ar = load_a_frag(Ar, NSRC, row0, k0, lane);
        v16h ai = load_a_frag(Ai, NSRC, row0, k0, lane);
        v16h bx = load_b_swz(Xs, NSRC, k0 >> 5, col0, lane);
        v16h by = load_b_swz(Ys, NSRC, k0 >> 5, col0, lane);
        cxr = WMMA_F16(ar, bx, cxr);
        cxi = WMMA_F16(ai, bx, cxi);
        cyr = WMMA_F16(ar, by, cyr);
        cyi = WMMA_F16(ai, by, cyi);
    }
    const int n = col0 + (lane & 15);
    const int h = lane >> 4;
#pragma unroll
    for (int v = 0; v < 8; ++v) {
        size_t o = (size_t)(row0 + v + 8 * h) * NSRC + n;
        Gxr[o] = (_Float16)cxr[v];
        Gxi[o] = (_Float16)cxi[v];
        Gyr[o] = (_Float16)cyr[v];
        Gyi[o] = (_Float16)cyi[v];
    }
}

// Stage 2: F = G * A_dft^T (complex*complex via 8 real accumulators), then per
// spectral bin apply the Wiener deconvolution using the analytic flipped-spectrum
// phase, and reduce sum(Re(Fd)) and sum(|Fd|^2) for this channel.
// 1024 waves: 32 x 32 tiles of the 512x512 (valid 511x511) spectrum, 8 WMMA/step.
__global__ void __launch_bounds__(256) k_stage2(
        const _Float16* __restrict__ Gxr, const _Float16* __restrict__ Gxi,
        const _Float16* __restrict__ Gyr, const _Float16* __restrict__ Gyi,
        const _Float16* __restrict__ Atrs, const _Float16* __restrict__ Atis,
        float* __restrict__ acc) {
    const int lane = threadIdx.x & 31;
    const int w = blockIdx.x * (blockDim.x >> 5) + (threadIdx.x >> 5);
    const int row0 = (w >> 5) * 16;   // k1 tile
    const int col0 = (w & 31) * 16;   // k2 tile
    v8f xrr = {}, xri = {}, xir = {}, xii = {};
    v8f yrr = {}, yri = {}, yir = {}, yii = {};
#pragma unroll
    for (int k0 = 0; k0 < NSRC; k0 += 32) {
        v16h gxr = load_a_frag(Gxr, NSRC, row0, k0, lane);
        v16h gxi = load_a_frag(Gxi, NSRC, row0, k0, lane);
        v16h gyr = load_a_frag(Gyr, NSRC, row0, k0, lane);
        v16h gyi = load_a_frag(Gyi, NSRC, row0, k0, lane);
        v16h br  = load_b_swz(Atrs, NPAD, k0 >> 5, col0, lane);
        v16h bi  = load_b_swz(Atis, NPAD, k0 >> 5, col0, lane);
        xrr = WMMA_F16(gxr, br, xrr);
        xri = WMMA_F16(gxr, bi, xri);
        xir = WMMA_F16(gxi, br, xir);
        xii = WMMA_F16(gxi, bi, xii);
        yrr = WMMA_F16(gyr, br, yrr);
        yri = WMMA_F16(gyr, bi, yri);
        yir = WMMA_F16(gyi, br, yir);
        yii = WMMA_F16(gyi, bi, yii);
    }
    float sRe = 0.0f, sMag = 0.0f;
    const int n = col0 + (lane & 15);
    const int h = lane >> 4;
#pragma unroll
    for (int v = 0; v < 8; ++v) {
        const int k1 = row0 + v + 8 * h;
        const int k2 = n;
        if (k1 < NFFT && k2 < NFFT) {
            // complex combine: F = (Gr + i Gi)(Br + i Bi)
            float Fxr = xrr[v] - xii[v], Fxi = xri[v] + xir[v];
            float Fyr = yrr[v] - yii[v], Fyi = yri[v] + yir[v];
            // FFT(flip(x)) = e^{+2pi i (k1+k2)/511} * conj(Fx)
            int ps = k1 + k2;
            if (ps >= NFFT) ps -= NFFT;
            float th = (PI2 / (float)NFFT) * (float)ps;
            float pim, pre;
            sincosf(th, &pim, &pre);
            float fr = pre * Fxr + pim * Fxi;          // Re(Fxf)
            float fi = pim * Fxr - pre * Fxi;          // Im(Fxf)
            float nr = fr * Fyr - fi * Fyi + AW_EPS;   // Fccorr + eps
            float ni = fr * Fyi + fi * Fyr;
            float m2 = Fxr * Fxr + Fxi * Fxi;          // Fxf*Fx = p*|Fx|^2
            float dr = pre * m2 + AW_EPS;
            float di = pim * m2;
            float inv = 1.0f / (dr * dr + di * di);
            float qr = (nr * dr + ni * di) * inv;      // Fdconv
            float qi = (ni * dr - nr * di) * inv;
            sRe  += qr;
            sMag += qr * qr + qi * qi;
        }
    }
#pragma unroll
    for (int o = 16; o > 0; o >>= 1) {
        sRe  += __shfl_xor(sRe, o, 32);
        sMag += __shfl_xor(sMag, o, 32);
    }
    if (lane == 0) {
        atomicAdd(acc + 0, sRe);
        atomicAdd(acc + 1, sMag);
    }
}

// loss = 0.5 * sum_c (1 - Re(sum Fd)^2 / (N * sum |Fd|^2)),  N = 511*511
__global__ void k_final(const float* __restrict__ acc, float* __restrict__ out) {
    if (threadIdx.x == 0 && blockIdx.x == 0) {
        const float Nf = 261121.0f;  // 511*511
        float f = 0.0f;
        for (int c = 0; c < NCH; ++c) {
            float sRe = acc[2 * c + 0];
            float sM  = acc[2 * c + 1];
            float t = (sM > 0.0f) ? (sRe * sRe) / (Nf * sM) : 0.0f;
            f += 1.0f - t;
        }
        out[0] = 0.5f * f;
    }
}

// ---------------- launch ----------------
extern "C" void kernel_launch(void* const* d_in, const int* in_sizes, int n_in,
                              void* d_out, int out_size, void* d_ws, size_t ws_size,
                              hipStream_t stream) {
    (void)in_sizes; (void)n_in; (void)out_size; (void)ws_size;
    const float* recon  = (const float*)d_in[0];   // y
    const float* target = (const float*)d_in[1];   // x (mode='reverse')

    char* ws = (char*)d_ws;
    size_t off = 0;
    _Float16* Ar   = (_Float16*)(ws + off); off += SZ_A;
    _Float16* Ai   = (_Float16*)(ws + off); off += SZ_A;
    _Float16* Atrs = (_Float16*)(ws + off); off += SZ_AT;
    _Float16* Atis = (_Float16*)(ws + off); off += SZ_AT;
    _Float16* Xs   = (_Float16*)(ws + off); off += SZ_X;
    _Float16* Ys   = (_Float16*)(ws + off); off += SZ_X;
    _Float16* Gxr  = (_Float16*)(ws + off); off += SZ_G;
    _Float16* Gxi  = (_Float16*)(ws + off); off += SZ_G;
    _Float16* Gyr  = (_Float16*)(ws + off); off += SZ_G;
    _Float16* Gyi  = (_Float16*)(ws + off); off += SZ_G;
    float*    acc  = (float*)(ws + off);           // 48 * 2 floats

    k_zero<<<1, 128, 0, stream>>>(acc, 2 * NCH);
    k_twiddle<<<(NPAD * NSRC + 255) / 256, 256, 0, stream>>>(Ar, Ai, Atrs, Atis);

    for (int c = 0; c < NCH; ++c) {
        const float* xc = target + (size_t)c * NSRC * NSRC;
        const float* yc = recon  + (size_t)c * NSRC * NSRC;
        k_convert<<<(NSRC * NSRC + 255) / 256, 256, 0, stream>>>(xc, yc, Xs, Ys);
        // 512 waves / 8 waves-per-block
        k_stage1<<<64, 256, 0, stream>>>(Ar, Ai, Xs, Ys, Gxr, Gxi, Gyr, Gyi);
        // 1024 waves / 8 waves-per-block
        k_stage2<<<128, 256, 0, stream>>>(Gxr, Gxi, Gyr, Gyi, Atrs, Atis, acc + 2 * c);
    }
    k_final<<<1, 64, 0, stream>>>(acc, (float*)d_out);
}
